// MusicTransformer_59459527246145
// MI455X (gfx1250) — matrix operand order repaired and checked by reading, not compiled, not run on hardware
//
#include <hip/hip_runtime.h>
#include <hip/hip_bf16.h>

// ---------------- model constants ----------------
#define DMODEL 512
#define HEADS  8
#define DHEAD  64
#define DFF    2048
#define NLAYER 12
#define BBATCH 4
#define SEQ    1024
#define NROWS  (BBATCH * SEQ)   // 4096
#define VOCAB  10000

typedef _Float16 v8h  __attribute__((ext_vector_type(8)));
typedef _Float16 v16h __attribute__((ext_vector_type(16)));
typedef float    v8f  __attribute__((ext_vector_type(8)));
typedef unsigned int u32x4 __attribute__((ext_vector_type(4)));
typedef int          i32x4 __attribute__((ext_vector_type(4)));
typedef int          i32x8 __attribute__((ext_vector_type(8)));

// Build a v16h fragment from two contiguous 8-half runs (matches CDNA5 WMMA
// 16-bit A/B per-lane element layouts, ISA 7.12.2).
static __device__ inline v16h ld_pair(const _Float16* p0, const _Float16* p1) {
  v8h a = *(const v8h*)p0;
  v8h b = *(const v8h*)p1;
  v16h r;
#pragma unroll
  for (int i = 0; i < 8; ++i) { r[i] = a[i]; r[i + 8] = b[i]; }
  return r;
}

// LDS byte offset of a generic pointer to __shared__ (low 32 bits of the
// LDS-aperture generic address are the LDS offset).
static __device__ inline unsigned lds_off(const void* p) {
  return (unsigned)(unsigned long long)(uintptr_t)p;
}

// Async global->LDS copy of 32 bytes per lane (2 x b128; instruction offset
// applies to BOTH the global and LDS addresses). Tracked by ASYNCcnt.
static __device__ inline void async_copy_32B(unsigned ldst, const void* gsrc) {
  asm volatile(
      "global_load_async_to_lds_b128 %0, %1, off\n\t"
      "global_load_async_to_lds_b128 %0, %1, off offset:16"
      :: "v"(ldst), "v"(gsrc) : "memory");
}
static __device__ inline void wait_asynccnt0() {
  asm volatile("s_wait_asynccnt 0x0" ::: "memory");
}
static __device__ inline void wait_tensorcnt0() {
#if __has_builtin(__builtin_amdgcn_s_wait_tensorcnt)
  __builtin_amdgcn_s_wait_tensorcnt((short)0);
#else
  asm volatile("s_wait_tensorcnt 0x0" ::: "memory");
#endif
}

// Reductions across the 16 lanes of each wave32 half (rows live per-half).
static __device__ inline float red_max16(float v) {
#pragma unroll
  for (int m = 1; m < 16; m <<= 1) v = fmaxf(v, __shfl_xor(v, m, 32));
  return v;
}
static __device__ inline float red_sum16(float v) {
#pragma unroll
  for (int m = 1; m < 16; m <<= 1) v += __shfl_xor(v, m, 32);
  return v;
}

// ---------------- embedding: h = word_emb[tok] + pos_emb[s] ----------------
__global__ __launch_bounds__(128) void embed_kernel(
    const int* __restrict__ data, const float* __restrict__ word_emb,
    const float* __restrict__ pos_emb, float* __restrict__ h,
    _Float16* __restrict__ x16) {
  const int row = blockIdx.x;            // b*SEQ + s
  const int s   = row & (SEQ - 1);
  const int tok = data[row];
  for (int c = threadIdx.x; c < DMODEL; c += 128) {
    const float v = word_emb[(size_t)tok * DMODEL + c] + pos_emb[(size_t)s * DMODEL + c];
    h[(size_t)row * DMODEL + c] = v;
    x16[(size_t)row * DMODEL + c] = (_Float16)v;
  }
}

// ---------------- generic GEMM: C[M,N] = A[M,K](f16) @ W[N,K]^T(f32) --------
// 128x128x32 block, 8 waves (2x4), each wave 64x32 via 4x2 v_wmma tiles.
// A tile staged via global_load_async_to_lds (ASYNCcnt path, no VGPR bounce);
// W tile goes through VGPRs for the f32->f16 convert.
__global__ __launch_bounds__(256) void gemm_xwT(
    const _Float16* __restrict__ A, const float* __restrict__ W,
    const float* __restrict__ bias, float* __restrict__ Cf,
    _Float16* __restrict__ Ch, int M, int N, int K, int relu) {
  __shared__ _Float16 As[128 * 40];
  __shared__ _Float16 Ws[128 * 40];

  const int t    = threadIdx.x;
  const int lane = t & 31;
  const int wv   = t >> 5;
  const int wm   = wv >> 2;          // 0..1 (M dir)
  const int wn   = wv & 3;           // 0..3 (N dir)
  const int ln   = lane & 15;
  const int hi   = lane >> 4;
  const int hi8  = hi * 8;
  const int hi16 = hi * 16;
  const int m0   = blockIdx.x * 128;
  const int n0   = blockIdx.y * 128;

  const int lr = t >> 1;             // 0..127 tile row
  const int lc = (t & 1) * 16;       // 0/16 K sub-col

  const unsigned a_lds = lds_off(As + lr * 40 + lc);

  const v8f z8 = {0.f, 0.f, 0.f, 0.f, 0.f, 0.f, 0.f, 0.f};
  v8f acc[4][2];
#pragma unroll
  for (int i = 0; i < 4; ++i)
#pragma unroll
    for (int j = 0; j < 2; ++j) acc[i][j] = z8;

  for (int k0 = 0; k0 < K; k0 += 32) {
    // A tile: async global->LDS (f16, no conversion needed)
    async_copy_32B(a_lds, A + (size_t)(m0 + lr) * K + k0 + lc);

    // W tile: f32 -> f16 convert on the fly through VGPRs
    {
      const int gn = n0 + lr;
      float wvv[16];
      if (gn < N) {
        const float4* gw = (const float4*)(W + (size_t)gn * K + k0 + lc);
#pragma unroll
        for (int q = 0; q < 4; ++q) {
          const float4 f = gw[q];
          wvv[q * 4 + 0] = f.x; wvv[q * 4 + 1] = f.y;
          wvv[q * 4 + 2] = f.z; wvv[q * 4 + 3] = f.w;
        }
        if (k0 + 32 < K)  // prefetch next K-slab of the weight stream
          __builtin_prefetch(W + (size_t)gn * K + k0 + 32 + lc, 0, 1);
      } else {
#pragma unroll
        for (int q = 0; q < 16; ++q) wvv[q] = 0.f;
      }
      v8h h0, h1;
#pragma unroll
      for (int q = 0; q < 8; ++q) {
        h0[q] = (_Float16)wvv[q];
        h1[q] = (_Float16)wvv[8 + q];
      }
      *(v8h*)(Ws + lr * 40 + lc)     = h0;
      *(v8h*)(Ws + lr * 40 + lc + 8) = h1;
    }
    wait_asynccnt0();          // A tile landed in LDS
    __syncthreads();

    v16h af[4], bf[2];
#pragma unroll
    for (int i = 0; i < 4; ++i) {
      const _Float16* ap = As + (wm * 64 + i * 16 + ln) * 40 + hi8;
      af[i] = ld_pair(ap, ap + 16);           // K runs {hi8..+7, hi8+16..+23}
    }
#pragma unroll
    for (int j = 0; j < 2; ++j) {
      const _Float16* bp = Ws + (wn * 32 + j * 16 + ln) * 40 + hi8;
      bf[j] = ld_pair(bp, bp + 16);           // K runs {hi8..+7, hi8+16..+23}
    }
#pragma unroll
    for (int i = 0; i < 4; ++i)
#pragma unroll
      for (int j = 0; j < 2; ++j)
        acc[i][j] = __builtin_amdgcn_wmma_f32_16x16x32_f16(
            false, af[i], false, bf[j], (short)0, acc[i][j], false, false);
    __syncthreads();
  }

  // epilogue: C layout -> lanes 0-15: M=r, lanes 16-31: M=8+r; N=lane%16
#pragma unroll
  for (int i = 0; i < 4; ++i) {
#pragma unroll
    for (int j = 0; j < 2; ++j) {
      const int nn = n0 + wn * 32 + j * 16 + ln;
      float bv = 0.f;
      if (bias != nullptr && nn < N) bv = bias[nn];
#pragma unroll
      for (int r = 0; r < 8; ++r) {
        const int mm = m0 + wm * 64 + i * 16 + hi8 + r;
        float v = acc[i][j][r] + bv;
        if (relu) v = fmaxf(v, 0.f);
        if (nn < N) {
          if (Cf) Cf[(size_t)mm * N + nn] = v;
          if (Ch) Ch[(size_t)mm * N + nn] = (_Float16)v;
        }
      }
    }
  }
}

// ---------------- flash attention (causal, scale=1/8) ----------------------
// grid: (SEQ/64, HEADS, BBATCH); 128 threads = 4 waves, 16 q-rows per wave.
// V tile staged by the Tensor Data Mover (TENSORcnt) when the builtin exists:
// 64x64 f16 tile, row stride 1024 elems, TDM pad 4 DWORDs / 32 DWORDs
// reproduces the 72-half padded LDS stride.
__global__ __launch_bounds__(128) void attn_fwd(
    const _Float16* __restrict__ q16, const _Float16* __restrict__ kv16,
    _Float16* __restrict__ o16) {
  __shared__ _Float16 Vs[64 * 72];   // V tile [key][d], shared by block
  __shared__ _Float16 Ps[64 * 72];   // P tiles, 16 rows per wave

  const int qblk = blockIdx.x;
  const int head = blockIdx.y;
  const int b    = blockIdx.z;
  const int t    = threadIdx.x;
  const int lane = t & 31;
  const int w    = t >> 5;
  const int ln   = lane & 15;
  const int hi   = lane >> 4;
  const int hi8  = hi * 8;
  const int hi16 = hi * 16;
  const int q0   = qblk * 64;
  const float scale = 0.125f;        // 1/sqrt(64)

  // Q A-fragments (per wave rows q0+w*16 .. +15), loaded once
  v16h qf[2];
#pragma unroll
  for (int ks = 0; ks < 2; ++ks) {
    const _Float16* qp = q16 + (size_t)(b * SEQ + q0 + w * 16 + ln) * DMODEL +
                         head * DHEAD + ks * 32 + hi8;
    qf[ks] = ld_pair(qp, qp + 16);
  }

  const v8f z8 = {0.f, 0.f, 0.f, 0.f, 0.f, 0.f, 0.f, 0.f};
  v8f O[4];
#pragma unroll
  for (int n4 = 0; n4 < 4; ++n4) O[n4] = z8;
  float mrow[8], lrow[8];
#pragma unroll
  for (int r = 0; r < 8; ++r) { mrow[r] = -1e30f; lrow[r] = 0.f; }

  for (int kb = 0; kb <= qblk; ++kb) {
    // ---- stage V tile [64 keys x 64 d] into LDS ----
#if __has_builtin(__builtin_amdgcn_tensor_load_to_lds)
    if (w == 0) {   // one wave issues the TDM descriptor (uniform values)
      const _Float16* gp = kv16 + (size_t)(b * SEQ + kb * 64) * 1024 +
                           512 + head * DHEAD;
      const unsigned long long ga = (unsigned long long)(uintptr_t)gp;
      u32x4 g0;
      g0[0] = 1u;                              // count=1, user descriptor
      g0[1] = lds_off(Vs);                     // lds_addr
      g0[2] = (unsigned)(ga & 0xffffffffu);    // global_addr[31:0]
      g0[3] = (unsigned)((ga >> 32) & 0x1ffffffu) | 0x80000000u;  // type=2
      i32x8 g1;
      // data_size=2B | pad_enable | pad_interval=32dw | pad_amount=4dw
      g1[0] = (int)0x07110000u;
      g1[1] = (int)(1024u << 16);              // tensor_dim0 = 1024 (low16)
      g1[2] = (int)(4096u << 16);              // dim0 hi16=0 | tensor_dim1 lo16
      g1[3] = (int)(64u << 16);                // dim1 hi16=0 | tile_dim0 = 64
      g1[4] = 64;                              // tile_dim1 = 64, tile_dim2 = 0
      g1[5] = 1024;                            // tensor_dim0_stride lo32
      g1[6] = 0;
      g1[7] = 0;
      i32x4 zz = {0, 0, 0, 0};
#if __clang_major__ >= 23
      i32x8 zz8 = {0, 0, 0, 0, 0, 0, 0, 0};
      __builtin_amdgcn_tensor_load_to_lds(g0, g1, zz, zz, zz8, 0);
#else
      __builtin_amdgcn_tensor_load_to_lds(g0, g1, zz, zz, 0);
#endif
      wait_tensorcnt0();
    }
#else
    {
      const int vr = t >> 1;
      const int vc = (t & 1) * 32;
      const _Float16* gp = kv16 + (size_t)(b * SEQ + kb * 64 + vr) * 1024 +
                           512 + head * DHEAD + vc;
      _Float16* lp = Vs + vr * 72 + vc;
#pragma unroll
      for (int i = 0; i < 4; ++i) *(v8h*)(lp + i * 8) = *(const v8h*)(gp + i * 8);
    }
#endif
    __syncthreads();

    // scores: S = Q K^T  (K B-fragments straight from global)
    v8f sc[4];
#pragma unroll
    for (int n = 0; n < 4; ++n) {
      sc[n] = z8;
#pragma unroll
      for (int ks = 0; ks < 2; ++ks) {
        const _Float16* kp = kv16 + (size_t)(b * SEQ + kb * 64 + n * 16 + ln) * 1024 +
                             head * DHEAD + ks * 32 + hi8;
        v16h kf = ld_pair(kp, kp + 16);
        sc[n] = __builtin_amdgcn_wmma_f32_16x16x32_f16(
            false, qf[ks], false, kf, (short)0, sc[n], false, false);
      }
    }

    // online softmax (rows live on 16-lane halves)
    float alpha[8];
#pragma unroll
    for (int r = 0; r < 8; ++r) {
      const int qrow = q0 + w * 16 + hi8 + r;
      float mx = -1e30f;
#pragma unroll
      for (int n = 0; n < 4; ++n) {
        const int key = kb * 64 + n * 16 + ln;
        float v = sc[n][r] * scale;
        if (key > qrow) v = -1e30f;   // causal mask
        sc[n][r] = v;
        mx = fmaxf(mx, v);
      }
      mx = red_max16(mx);
      const float mn = fmaxf(mrow[r], mx);
      const float al = __expf(mrow[r] - mn);
      float rs = 0.f;
#pragma unroll
      for (int n = 0; n < 4; ++n) {
        const float p = __expf(sc[n][r] - mn);
        rs += p;
        Ps[(w * 16 + hi8 + r) * 72 + n * 16 + ln] = (_Float16)p;  // C->LDS
      }
      rs = red_sum16(rs);
      lrow[r] = lrow[r] * al + rs;
      mrow[r] = mn;
      alpha[r] = al;
    }
#pragma unroll
    for (int n4 = 0; n4 < 4; ++n4)
#pragma unroll
      for (int r = 0; r < 8; ++r) O[n4][r] *= alpha[r];

    asm volatile("s_wait_dscnt 0" ::: "memory");  // own-wave Ps stores visible

    // O += P V  (P re-read in A layout, V columns from LDS in B layout)
#pragma unroll
    for (int ks = 0; ks < 2; ++ks) {
      const _Float16* pp = Ps + (w * 16 + ln) * 72 + ks * 32 + hi16;
      v16h pf = ld_pair(pp, pp + 8);
#pragma unroll
      for (int n4 = 0; n4 < 4; ++n4) {
        v16h vf;
#pragma unroll
        for (int e = 0; e < 16; ++e)
          vf[e] = Vs[(ks * 32 + hi16 + e) * 72 + n4 * 16 + ln];
        O[n4] = __builtin_amdgcn_wmma_f32_16x16x32_f16(
            false, pf, false, vf, (short)0, O[n4], false, false);
      }
    }
    __syncthreads();  // protect Vs for next kb
  }

  // normalize and write f16 output [row, head*64 + d]
#pragma unroll
  for (int n4 = 0; n4 < 4; ++n4) {
#pragma unroll
    for (int r = 0; r < 8; ++r) {
      const float val = O[n4][r] / lrow[r];
      const int qg = b * SEQ + q0 + w * 16 + hi8 + r;
      o16[(size_t)qg * DMODEL + head * DHEAD + n4 * 16 + ln] = (_Float16)val;
    }
  }
}

// ---------------- residual add + LayerNorm -> h(f32), x16(f16) -------------
__global__ __launch_bounds__(128) void residual_ln(
    const float* __restrict__ hin, const float* __restrict__ proj,
    const float* __restrict__ g, const float* __restrict__ beta,
    float* __restrict__ hout, _Float16* __restrict__ x16) {
  __shared__ float rbuf[8];
  const int row  = blockIdx.x;
  const int t    = threadIdx.x;
  const int lane = t & 31;
  const int wid  = t >> 5;

  float vals[4];
  float s = 0.f, s2 = 0.f;
#pragma unroll
  for (int i = 0; i < 4; ++i) {
    const int c = t + i * 128;
    const float v = hin[(size_t)row * DMODEL + c] + proj[(size_t)row * DMODEL + c];
    vals[i] = v; s += v; s2 += v * v;
  }
#pragma unroll
  for (int m = 1; m < 32; m <<= 1) {
    s  += __shfl_xor(s, m, 32);
    s2 += __shfl_xor(s2, m, 32);
  }
  if (lane == 0) { rbuf[wid] = s; rbuf[4 + wid] = s2; }
  __syncthreads();
  s  = rbuf[0] + rbuf[1] + rbuf[2] + rbuf[3];
  s2 = rbuf[4] + rbuf[5] + rbuf[6] + rbuf[7];
  const float mean = s * (1.f / DMODEL);
  const float var  = s2 * (1.f / DMODEL) - mean * mean;
  const float rstd = rsqrtf(var + 1e-5f);
#pragma unroll
  for (int i = 0; i < 4; ++i) {
    const int c = t + i * 128;
    const float y = (vals[i] - mean) * rstd * g[c] + beta[c];
    hout[(size_t)row * DMODEL + c] = y;
    x16[(size_t)row * DMODEL + c] = (_Float16)y;
  }
}

// ---------------- host driver ----------------------------------------------
extern "C" void kernel_launch(void* const* d_in, const int* in_sizes, int n_in,
                              void* d_out, int out_size, void* d_ws, size_t ws_size,
                              hipStream_t stream) {
  (void)in_sizes; (void)n_in; (void)out_size; (void)ws_size;
  const int*   data     = (const int*)  d_in[0];
  const float* word_emb = (const float*)d_in[1];
  const float* pos_emb  = (const float*)d_in[2];
  const float* Wq       = (const float*)d_in[3];
  const float* Wkv      = (const float*)d_in[4];
  const float* Wo       = (const float*)d_in[5];
  const float* ln1_g    = (const float*)d_in[6];
  const float* ln1_b    = (const float*)d_in[7];
  const float* W1       = (const float*)d_in[8];
  const float* b1       = (const float*)d_in[9];
  const float* W2       = (const float*)d_in[10];
  const float* b2       = (const float*)d_in[11];
  const float* ln2_g    = (const float*)d_in[12];
  const float* ln2_b    = (const float*)d_in[13];
  const float* out_b    = (const float*)d_in[14];
  float* out = (float*)d_out;

  char* ws = (char*)d_ws;
  float*    h      = (float*)   (ws + 0);                 //  8 MB f32 [4096,512]
  _Float16* x16    = (_Float16*)(ws + 8388608);           //  4 MB f16 [4096,512]
  _Float16* q16    = (_Float16*)(ws + 12582912);          //  4 MB f16 [4096,512]
  _Float16* kv16   = (_Float16*)(ws + 16777216);          //  8 MB f16 [4096,1024]
  _Float16* attn16 = (_Float16*)(ws + 25165824);          //  4 MB f16 [4096,512]
  _Float16* ff16   = (_Float16*)(ws + 29360128);          // 16 MB f16 [4096,2048]
  float*    tmp    = (float*)   (ws + 46137344);          //  8 MB f32 [4096,512]

  embed_kernel<<<NROWS, 128, 0, stream>>>(data, word_emb, pos_emb, h, x16);

  for (int l = 0; l < NLAYER; ++l) {
    const float* wq  = Wq  + (size_t)l * DMODEL * DMODEL;
    const float* wkv = Wkv + (size_t)l * 2 * DMODEL * DMODEL;
    const float* wo  = Wo  + (size_t)l * DMODEL * DMODEL;
    const float* w1  = W1  + (size_t)l * DFF * DMODEL;
    const float* w2  = W2  + (size_t)l * DMODEL * DFF;

    gemm_xwT<<<dim3(NROWS / 128, DMODEL / 128), 256, 0, stream>>>(
        x16, wq, nullptr, nullptr, q16, NROWS, DMODEL, DMODEL, 0);
    gemm_xwT<<<dim3(NROWS / 128, (2 * DMODEL) / 128), 256, 0, stream>>>(
        x16, wkv, nullptr, nullptr, kv16, NROWS, 2 * DMODEL, DMODEL, 0);
    attn_fwd<<<dim3(SEQ / 64, HEADS, BBATCH), 128, 0, stream>>>(q16, kv16, attn16);
    gemm_xwT<<<dim3(NROWS / 128, DMODEL / 128), 256, 0, stream>>>(
        attn16, wo, nullptr, tmp, nullptr, NROWS, DMODEL, DMODEL, 0);
    residual_ln<<<NROWS, 128, 0, stream>>>(h, tmp, ln1_g + l * DMODEL,
                                           ln1_b + l * DMODEL, h, x16);
    gemm_xwT<<<dim3(NROWS / 128, DFF / 128), 256, 0, stream>>>(
        x16, w1, b1 + (size_t)l * DFF, nullptr, ff16, NROWS, DFF, DMODEL, 1);
    gemm_xwT<<<dim3(NROWS / 128, DMODEL / 128), 256, 0, stream>>>(
        ff16, w2, b2 + (size_t)l * DMODEL, tmp, nullptr, NROWS, DMODEL, DFF, 0);
    residual_ln<<<NROWS, 128, 0, stream>>>(h, tmp, ln2_g + l * DMODEL,
                                           ln2_b + l * DMODEL, h, x16);
  }

  // logits = h @ word_emb^T + out_b  (tied projection), f32 output
  gemm_xwT<<<dim3(NROWS / 128, (VOCAB + 127) / 128), 256, 0, stream>>>(
      x16, word_emb, out_b, out, nullptr, NROWS, VOCAB, DMODEL, 0);
}
